// EncoderLayer_49151605736114
// MI455X (gfx1250) — compile-verified, matrix operand order checked by reading, not run on hardware
//
#include <hip/hip_runtime.h>
#include <hip/hip_bf16.h>

typedef __attribute__((ext_vector_type(16))) __bf16 v16bf;
typedef __attribute__((ext_vector_type(8)))  float  v8f;
typedef __attribute__((ext_vector_type(4)))  unsigned int v4u;
typedef __attribute__((ext_vector_type(8)))  int v8i;
typedef __attribute__((ext_vector_type(4)))  int v4i;

#define B_   4
#define S_   2048
#define D_   1024
#define H_   16
#define E_   8
#define F_   4096
#define DH_  64
#define N_   (B_ * S_)
#define CAP_ 1280   /* int(1.25 * 8192 / 8) */

// ---------------------------------------------------------------------------
// WMMA helpers
// ---------------------------------------------------------------------------
__device__ inline v8f wmma_bf16(v16bf a, v16bf b, v8f c) {
    return __builtin_amdgcn_wmma_f32_16x16x32_bf16(
        /*neg_a=*/false, a, /*neg_b=*/false, b,
        /*c_mod=*/(short)0, c, /*reuse_a=*/false, /*reuse_b=*/false);
}

// Build a 16x32 fragment from an LDS row (A layout / row-staged-B layout).
// Lanes 0-15: K=0..7 / 16..23; lanes 16-31: K=8..15 / 24..31.
__device__ inline v16bf lds_frag(const __bf16* row, int h) {
    union { v16bf v; uint4 u[2]; } f;
    f.u[0] = *(const uint4*)(row + h * 8);
    f.u[1] = *(const uint4*)(row + 16 + h * 8);
    return f.v;
}

// ---------------------------------------------------------------------------
// CDNA5 async data movement helpers
// ---------------------------------------------------------------------------
// Per-lane 16B global -> LDS async copy (ASYNCcnt tracked).
__device__ inline void async_copy_b128(unsigned lds_off, const void* gptr) {
    asm volatile("global_load_async_to_lds_b128 %0, %1, off"
                 :: "v"(lds_off), "v"(gptr) : "memory");
}
__device__ inline void wait_asynccnt0() {
    asm volatile("s_wait_asynccnt 0x0" ::: "memory");
}

// Tensor Data Mover: 2D tile (tile_x elems of 2B) x tile_y rows, global row
// stride row_stride elems, into LDS at lds_off with 4-DWORD padding every
// 16 DWORDs (=> 40-element padded rows for tile_x==32).
__device__ inline void tdm_load_tile_2d(unsigned lds_off, const void* gaddr,
                                        unsigned tile_x, unsigned tile_y,
                                        unsigned long long row_stride) {
    unsigned long long ga = (unsigned long long)gaddr;
    v4u g0;
    g0[0] = 1u;                                          // count=1, user D#
    g0[1] = lds_off;                                     // lds_addr
    g0[2] = (unsigned)(ga & 0xffffffffu);                // global_addr[31:0]
    g0[3] = (unsigned)((ga >> 32) & 0x01ffffffu)         // global_addr[56:32]
            | (2u << 30);                                // type=2 (image)
    const unsigned td0 = (unsigned)row_stride;           // tensor_dim0
    const unsigned td1 = 0x7fffffffu;                    // tensor_dim1 (big)
    v8i g1;
    g1[0] = (int)((1u << 16)      // data_size = 1 (2 bytes)
                | (1u << 20)      // pad_enable
                | (3u << 22)      // pad_interval: 16 DWORDs
                | (3u << 25));    // pad_amount:   4 DWORDs
    g1[1] = (int)((td0 & 0xffffu) << 16);                         // td0 lo
    g1[2] = (int)((td0 >> 16) | ((td1 & 0xffffu) << 16));         // td0 hi|td1 lo
    g1[3] = (int)((td1 >> 16) | (tile_x << 16));                  // td1 hi|tile0
    g1[4] = (int)(tile_y & 0xffffu);                              // tile1, tile2=0
    g1[5] = (int)(unsigned)(row_stride & 0xffffffffu);            // dim0_stride lo
    g1[6] = (int)(unsigned)((row_stride >> 32) & 0xffffu);        // dim0_stride hi
    g1[7] = 0;
    v4i z4 = {0, 0, 0, 0};
    v8i z8 = {0, 0, 0, 0, 0, 0, 0, 0};
    __builtin_amdgcn_tensor_load_to_lds(g0, g1, z4, z4, z8, 0);
}

// ---------------------------------------------------------------------------
// fp32 -> bf16 conversion / zero fills
// ---------------------------------------------------------------------------
__global__ void k_cvt(const float* __restrict__ in, __bf16* __restrict__ out,
                      long long n) {
    long long i = (long long)blockIdx.x * blockDim.x + threadIdx.x;
    long long stride = (long long)gridDim.x * blockDim.x;
    for (; i < n; i += stride) out[i] = (__bf16)in[i];
}

__global__ void k_zero16(uint4* __restrict__ p, long long n16) {
    long long i = (long long)blockIdx.x * blockDim.x + threadIdx.x;
    long long stride = (long long)gridDim.x * blockDim.x;
    uint4 z; z.x = z.y = z.z = z.w = 0u;
    for (; i < n16; i += stride) p[i] = z;
}

__global__ void k_zerof(float* __restrict__ p, int n) {
    int i = threadIdx.x;
    if (i < n) p[i] = 0.f;
}

// ---------------------------------------------------------------------------
// LayerNorm: one wave per row of D=1024, bf16 output
// ---------------------------------------------------------------------------
__global__ __launch_bounds__(256)
void k_ln(const float* __restrict__ x, const float* __restrict__ g,
          const float* __restrict__ bt, __bf16* __restrict__ out) {
    const int lane = threadIdx.x & 31, wid = threadIdx.x >> 5;
    const long long row = (long long)blockIdx.x * 8 + wid;
    const float* xr = x + row * D_;
    float4 v[8];
    float s = 0.f, ss = 0.f;
#pragma unroll
    for (int j = 0; j < 8; ++j) {
        v[j] = *(const float4*)&xr[j * 128 + lane * 4];
        s  += v[j].x + v[j].y + v[j].z + v[j].w;
        ss += v[j].x * v[j].x + v[j].y * v[j].y + v[j].z * v[j].z + v[j].w * v[j].w;
    }
#pragma unroll
    for (int off = 1; off < 32; off <<= 1) {
        s  += __shfl_xor(s, off, 32);
        ss += __shfl_xor(ss, off, 32);
    }
    const float mean = s * (1.f / D_);
    const float var  = ss * (1.f / D_) - mean * mean;
    const float rstd = rsqrtf(var + 1e-5f);
#pragma unroll
    for (int j = 0; j < 8; ++j) {
        int c = j * 128 + lane * 4;
        float4 gg = *(const float4*)&g[c];
        float4 bb = *(const float4*)&bt[c];
        __bf16* o = out + row * D_ + c;
        o[0] = (__bf16)((v[j].x - mean) * rstd * gg.x + bb.x);
        o[1] = (__bf16)((v[j].y - mean) * rstd * gg.y + bb.y);
        o[2] = (__bf16)((v[j].z - mean) * rstd * gg.z + bb.z);
        o[3] = (__bf16)((v[j].w - mean) * rstd * gg.w + bb.w);
    }
}

// ---------------------------------------------------------------------------
// Generic bf16 WMMA GEMM: C[M,N] = A[M,K] * B(+bias, epilogue)
//   transB=1: B stored [N,K];  transB=0: B stored [K,N]
// EPI: 0 = fp32+bias  1 = fp32+bias+residual  2 = bf16+bias  3 = bf16+bias+ReLU
// Macro tile 64(M) x 128(N), K-step 32. 8 waves in 2x4, 32x32 C per wave.
// A tile fetched by the Tensor Data Mover; B tile by async LDS copies.
// ---------------------------------------------------------------------------
template <int EPI>
__global__ __launch_bounds__(256)
void k_gemm(const __bf16* __restrict__ A, const __bf16* __restrict__ Bm,
            const float* __restrict__ bias, const float* __restrict__ resid,
            void* __restrict__ Cout, int M, int N, int K, int transB,
            long long sA, long long sB, long long sBias, long long sC) {
    __shared__ __align__(16) __bf16 As[64][40];    // [m][k], 80B rows
    __shared__ __align__(16) __bf16 Bs[128][40];   // [n][k] (pre-transposed)

    const int tid = threadIdx.x, lane = tid & 31, wid = tid >> 5;
    const int h = lane >> 4, l16 = lane & 15;
    const int z = blockIdx.z;
    A += (long long)z * sA;
    Bm += (long long)z * sB;
    bias += (long long)z * sBias;
    float*  Cf = (float*)Cout + (long long)z * sC;
    __bf16* Ch = (__bf16*)Cout + (long long)z * sC;

    const int m0 = blockIdx.y * 64, n0 = blockIdx.x * 128;
    const int waveM = wid >> 2, waveN = wid & 3;
    const unsigned as_lds = (unsigned)(uintptr_t)&As[0][0];

    v8f acc[2][2] = {};

    for (int k0 = 0; k0 < K; k0 += 32) {
        // A tile 64x32 via TDM (wave 0 issues; pad lands rows at 40 elems)
        if (wid == 0) {
            tdm_load_tile_2d(as_lds, &A[(long long)m0 * K + k0], 32u, 64u,
                             (unsigned long long)K);
            __builtin_amdgcn_s_wait_tensorcnt(0);
        }
        // B tile into Bs[n][k]: 128x32 = 4096 elems
        if (transB) {
            for (int i = 0; i < 2; ++i) {
                int idx = tid + i * 256;
                int n = idx >> 2, kg = (idx & 3) * 8;
                const __bf16* src = &Bm[(long long)(n0 + n) * K + k0 + kg];
                async_copy_b128((unsigned)(uintptr_t)&Bs[n][kg], src);
                if (k0 + 32 < K) __builtin_prefetch(src + 32, 0, 1);
            }
            wait_asynccnt0();
        } else {
            for (int i = 0; i < 2; ++i) {
                int idx = tid + i * 256;
                int k = idx >> 4, ng = (idx & 15) * 8;
                union { uint4 u; __bf16 e[8]; } t;
                t.u = *(const uint4*)&Bm[(long long)(k0 + k) * N + n0 + ng];
#pragma unroll
                for (int j = 0; j < 8; ++j) Bs[ng + j][k] = t.e[j];
            }
        }
        __syncthreads();

        v16bf af[2], bfr[2];
#pragma unroll
        for (int mt = 0; mt < 2; ++mt)
            af[mt] = lds_frag(&As[waveM * 32 + mt * 16 + l16][0], h);
#pragma unroll
        for (int nt = 0; nt < 2; ++nt)
            bfr[nt] = lds_frag(&Bs[waveN * 32 + nt * 16 + l16][0], h);
#pragma unroll
        for (int mt = 0; mt < 2; ++mt)
#pragma unroll
            for (int nt = 0; nt < 2; ++nt)
                acc[mt][nt] = wmma_bf16(af[mt], bfr[nt], acc[mt][nt]);
        __syncthreads();
    }

    // Epilogue. C layout: VGPR r -> row (r + 8*(lane>>4)), col = lane&15.
#pragma unroll
    for (int mt = 0; mt < 2; ++mt)
#pragma unroll
        for (int nt = 0; nt < 2; ++nt)
#pragma unroll
            for (int r = 0; r < 8; ++r) {
                int row = m0 + waveM * 32 + mt * 16 + r + 8 * h;
                int col = n0 + waveN * 32 + nt * 16 + l16;
                float v = acc[mt][nt][r] + bias[col];
                long long o = (long long)row * N + col;
                if (EPI == 0) {
                    Cf[o] = v;
                } else if (EPI == 1) {
                    Cf[o] = v + resid[o];
                } else if (EPI == 2) {
                    Ch[o] = (__bf16)v;
                } else {
                    Ch[o] = (__bf16)fmaxf(v, 0.f);
                }
            }
}

// ---------------------------------------------------------------------------
// Flash attention. Block = 128 q rows for one (b, head); 8 waves x 16 rows.
// KV streamed in 32-row chunks; K via async LDS copies, V transposed in VALU.
// qkv layout: [B*S, 3*D]; q at col h*64, k at D + h*64, v at 2D + h*64.
// ---------------------------------------------------------------------------
__global__ __launch_bounds__(256)
void k_attn(const __bf16* __restrict__ qkv,
            const unsigned char* __restrict__ pad,
            __bf16* __restrict__ out) {
    __shared__ __align__(16) __bf16 Ks[32][72];       // [kv][d], 144B rows
    __shared__ __align__(16) __bf16 VsT[64][40];      // [d][kv]
    __shared__ __align__(16) __bf16 Ps[8][16][40];    // per-wave P scratch

    const int tid = threadIdx.x, lane = tid & 31, wid = tid >> 5;
    const int h2 = lane >> 4, l16 = lane & 15;
    const int hd = blockIdx.y;
    const int b  = blockIdx.z;
    const int q0 = blockIdx.x * 128;
    const long long ldq = 3 * D_;
    const __bf16* qb = qkv + (long long)(b * S_) * ldq;

    // Q fragments: 16 rows x 64 dims = 2 A-frags, held in registers
    v16bf qa[2];
    {
        int qrow = q0 + wid * 16 + l16;
        const __bf16* qp = qb + (long long)qrow * ldq + hd * DH_;
        union { v16bf v; uint4 u[2]; } f;
#pragma unroll
        for (int c = 0; c < 2; ++c) {
            f.u[0] = *(const uint4*)(qp + c * 32 + h2 * 8);
            f.u[1] = *(const uint4*)(qp + c * 32 + 16 + h2 * 8);
            qa[c] = f.v;
        }
    }

    float mrun[8], lrun[8];
    v8f oacc[4] = {};
#pragma unroll
    for (int r = 0; r < 8; ++r) { mrun[r] = -3.0e38f; lrun[r] = 0.f; }

    for (int kv = 0; kv < S_; kv += 32) {
        // Cooperative KV chunk load: K async -> Ks, V -> VsT transposed
        {
            int r = tid >> 3, dg = (tid & 7) * 8;
            const __bf16* base = qb + (long long)(kv + r) * ldq + hd * DH_ + dg;
            async_copy_b128((unsigned)(uintptr_t)&Ks[r][dg], base + D_);
            union { uint4 u; __bf16 e[8]; } t;
            t.u = *(const uint4*)(base + 2 * D_);
#pragma unroll
            for (int j = 0; j < 8; ++j) VsT[dg + j][r] = t.e[j];
            wait_asynccnt0();
        }
        __syncthreads();

        // scores: Q(16x64) . K^T(64x32) -> 2 C tiles, accumulate over d-chunks
        v8f sacc[2] = {};
#pragma unroll
        for (int c = 0; c < 2; ++c) {
#pragma unroll
            for (int nt = 0; nt < 2; ++nt) {
                v16bf kb;
                {
                    union { v16bf v; uint4 u[2]; } f;
                    const __bf16* kr = &Ks[nt * 16 + l16][c * 32];
                    f.u[0] = *(const uint4*)(kr + h2 * 8);
                    f.u[1] = *(const uint4*)(kr + 16 + h2 * 8);
                    kb = f.v;
                }
                sacc[nt] = wmma_bf16(qa[c], kb, sacc[nt]);
            }
        }

        const bool msk0 = pad[b * S_ + kv + l16] != 0;
        const bool msk1 = pad[b * S_ + kv + 16 + l16] != 0;

        // online softmax, row stats reduced over 16-lane column groups
#pragma unroll
        for (int r = 0; r < 8; ++r) {
            float s0 = msk0 ? -1e9f : sacc[0][r] * 0.125f;  // 1/sqrt(64)
            float s1 = msk1 ? -1e9f : sacc[1][r] * 0.125f;
            float cm = fmaxf(s0, s1);
#pragma unroll
            for (int off = 1; off < 16; off <<= 1)
                cm = fmaxf(cm, __shfl_xor(cm, off, 32));
            float mn = fmaxf(mrun[r], cm);
            float p0 = __expf(s0 - mn), p1 = __expf(s1 - mn);
            float rs = p0 + p1;
#pragma unroll
            for (int off = 1; off < 16; off <<= 1)
                rs += __shfl_xor(rs, off, 32);
            float alpha = __expf(mrun[r] - mn);
            lrun[r] = lrun[r] * alpha + rs;
            mrun[r] = mn;
#pragma unroll
            for (int dt = 0; dt < 4; ++dt) oacc[dt][r] *= alpha;
            Ps[wid][r + 8 * h2][l16]      = (__bf16)p0;
            Ps[wid][r + 8 * h2][16 + l16] = (__bf16)p1;
        }

        // P(16x32) . V(32x64): re-load P as an A-fragment from LDS scratch
        v16bf pa;
        {
            union { v16bf v; uint4 u[2]; } f;
            const __bf16* pr = &Ps[wid][l16][0];
            f.u[0] = *(const uint4*)(pr + h2 * 8);
            f.u[1] = *(const uint4*)(pr + 16 + h2 * 8);
            pa = f.v;
        }
#pragma unroll
        for (int dt = 0; dt < 4; ++dt) {
            v16bf vb;
            {
                union { v16bf v; uint4 u[2]; } f;
                const __bf16* vr = &VsT[dt * 16 + l16][0];
                f.u[0] = *(const uint4*)(vr + h2 * 8);
                f.u[1] = *(const uint4*)(vr + 16 + h2 * 8);
                vb = f.v;
            }
            oacc[dt] = wmma_bf16(pa, vb, oacc[dt]);
        }
        __syncthreads();
    }

    // normalize and write [B*S, D] bf16
#pragma unroll
    for (int dt = 0; dt < 4; ++dt)
#pragma unroll
        for (int r = 0; r < 8; ++r) {
            int qrow = q0 + wid * 16 + r + 8 * h2;
            float v = oacc[dt][r] / lrun[r];
            out[(long long)(b * S_ + qrow) * D_ + hd * DH_ + dt * 16 + l16] =
                (__bf16)v;
        }
}

// ---------------------------------------------------------------------------
// Router: one wave per token. stats: [0..7]=cnt, [8..15]=Psum, [16]=zsum,
// [17]=tmsum
// ---------------------------------------------------------------------------
__global__ __launch_bounds__(256)
void k_router(const __bf16* __restrict__ x2h, const float* __restrict__ rw,
              const float* __restrict__ rb, const unsigned char* __restrict__ pad,
              int* __restrict__ eidx, float* __restrict__ gate,
              float* __restrict__ stats) {
    const int lane = threadIdx.x & 31, wid = threadIdx.x >> 5;
    const int n = blockIdx.x * 8 + wid;
    float acc[8] = {};
    for (int j = 0; j < 32; ++j) {
        int d = j * 32 + lane;
        float xv = (float)x2h[(long long)n * D_ + d];
        const float* w = rw + d * 8;
        float4 a = *(const float4*)w;
        float4 c = *(const float4*)(w + 4);
        acc[0] += xv * a.x; acc[1] += xv * a.y;
        acc[2] += xv * a.z; acc[3] += xv * a.w;
        acc[4] += xv * c.x; acc[5] += xv * c.y;
        acc[6] += xv * c.z; acc[7] += xv * c.w;
    }
#pragma unroll
    for (int off = 1; off < 32; off <<= 1)
#pragma unroll
        for (int e = 0; e < 8; ++e) acc[e] += __shfl_xor(acc[e], off, 32);
    if (lane == 0) {
        float tm = pad[n] ? 0.f : 1.f;
        float mx = -3e38f;
        float lg[8];
#pragma unroll
        for (int e = 0; e < 8; ++e) { lg[e] = acc[e] + rb[e]; mx = fmaxf(mx, lg[e]); }
        float pr[8], sum = 0.f;
#pragma unroll
        for (int e = 0; e < 8; ++e) { pr[e] = __expf(lg[e] - mx); sum += pr[e]; }
        int best = 0;
#pragma unroll
        for (int e = 1; e < 8; ++e) if (pr[e] > pr[best]) best = e;
        float inv = 1.f / sum;
        gate[n] = pr[best] * inv;
        eidx[n] = best;
        float lse = mx + __logf(sum);
        atomicAdd(&stats[16], tm * lse * lse);
        atomicAdd(&stats[17], tm);
        atomicAdd(&stats[best], tm);
#pragma unroll
        for (int e = 0; e < 8; ++e) atomicAdd(&stats[8 + e], tm * pr[e] * inv);
    }
}

// ---------------------------------------------------------------------------
// Sequential capacity scan: single wave, ballot-based per-expert positions
// ---------------------------------------------------------------------------
__global__ void k_scan(const int* __restrict__ eidx,
                       const unsigned char* __restrict__ pad,
                       int* __restrict__ slot) {
    const int lane = threadIdx.x;
    int base[8] = {};
    unsigned below = (1u << lane) - 1u;
    for (int t = 0; t < N_; t += 32) {
        int i = t + lane;
        int e = eidx[i];
        bool valid = pad[i] == 0;
        int pos = -1;
#pragma unroll
        for (int ex = 0; ex < 8; ++ex) {
            unsigned m = (unsigned)__ballot(valid && (e == ex));
            if (valid && e == ex) pos = base[ex] + __popc(m & below);
            base[ex] += __popc(m);
        }
        slot[i] = (valid && pos < CAP_) ? (e * CAP_ + pos) : -1;
    }
}

// ---------------------------------------------------------------------------
// Dispatch scatter: one wave per token, 2KB row copy via uint4
// ---------------------------------------------------------------------------
__global__ __launch_bounds__(256)
void k_dispatch(const __bf16* __restrict__ x2h, const int* __restrict__ slot,
                __bf16* __restrict__ disp) {
    const int lane = threadIdx.x & 31, wid = threadIdx.x >> 5;
    const int i = blockIdx.x * 8 + wid;
    int sl = slot[i];
    if (sl < 0) return;
    const uint4* src = (const uint4*)(x2h + (long long)i * D_);
    uint4* dst = (uint4*)(disp + (long long)sl * D_);
#pragma unroll
    for (int j = 0; j < 4; ++j) dst[lane + j * 32] = src[lane + j * 32];
}

// ---------------------------------------------------------------------------
// Final gather: src2 = src1 + keep * gate * o[slot]
// ---------------------------------------------------------------------------
__global__ __launch_bounds__(256)
void k_final(const float* __restrict__ src1, const float* __restrict__ oexp,
             const int* __restrict__ slot, const float* __restrict__ gate,
             float* __restrict__ out) {
    const int lane = threadIdx.x & 31, wid = threadIdx.x >> 5;
    const int i = blockIdx.x * 8 + wid;
    int sl = slot[i];
    float gv = (sl >= 0) ? gate[i] : 0.f;
#pragma unroll
    for (int j = 0; j < 8; ++j) {
        int c = j * 128 + lane * 4;
        float4 r = *(const float4*)&src1[(long long)i * D_ + c];
        if (sl >= 0) {
            float4 ov = *(const float4*)&oexp[(long long)sl * D_ + c];
            r.x += gv * ov.x; r.y += gv * ov.y;
            r.z += gv * ov.z; r.w += gv * ov.w;
        }
        *(float4*)&out[(long long)i * D_ + c] = r;
    }
}

__global__ void k_loss(const float* __restrict__ stats, float* __restrict__ out) {
    if (threadIdx.x == 0) {
        float denom = fmaxf(stats[17], 1.f);
        float lb = 0.f;
        for (int e = 0; e < 8; ++e)
            lb += (stats[e] / denom) * (stats[8 + e] / denom);
        out[0] = (float)E_ * lb;
        out[1] = stats[16] / denom;
    }
}

// ---------------------------------------------------------------------------
// Host orchestration
// ---------------------------------------------------------------------------
extern "C" void kernel_launch(void* const* d_in, const int* in_sizes, int n_in,
                              void* d_out, int out_size, void* d_ws,
                              size_t ws_size, hipStream_t stream) {
    const float* src          = (const float*)d_in[0];
    const unsigned char* pad  = (const unsigned char*)d_in[1];
    const float* ln1_g        = (const float*)d_in[2];
    const float* ln1_b        = (const float*)d_in[3];
    const float* in_proj_w    = (const float*)d_in[4];
    const float* in_proj_b    = (const float*)d_in[5];
    const float* out_proj_w   = (const float*)d_in[6];
    const float* out_proj_b   = (const float*)d_in[7];
    const float* ln2_g        = (const float*)d_in[8];
    const float* ln2_b        = (const float*)d_in[9];
    const float* router_w     = (const float*)d_in[10];
    const float* router_b     = (const float*)d_in[11];
    const float* w1           = (const float*)d_in[12];
    const float* b1           = (const float*)d_in[13];
    const float* w2           = (const float*)d_in[14];
    const float* b2           = (const float*)d_in[15];
    float* out                = (float*)d_out;

    char* ws = (char*)d_ws;
    size_t off = 0;
    auto alloc = [&](size_t bytes) -> void* {
        void* p = ws + off;
        off = (off + bytes + 255) & ~(size_t)255;
        return p;
    };
    __bf16* xh    = (__bf16*)alloc((size_t)N_ * D_ * 2);
    __bf16* qkvh  = (__bf16*)alloc((size_t)N_ * 3 * D_ * 2);
    __bf16* attnh = (__bf16*)alloc((size_t)N_ * D_ * 2);
    float*  src1  = (float*) alloc((size_t)N_ * D_ * 4);
    __bf16* x2h   = (__bf16*)alloc((size_t)N_ * D_ * 2);
    __bf16* wqkv  = (__bf16*)alloc((size_t)3 * D_ * D_ * 2);
    __bf16* wout  = (__bf16*)alloc((size_t)D_ * D_ * 2);
    __bf16* w1h   = (__bf16*)alloc((size_t)E_ * D_ * F_ * 2);
    __bf16* w2h   = (__bf16*)alloc((size_t)E_ * F_ * D_ * 2);
    __bf16* disp  = (__bf16*)alloc((size_t)E_ * CAP_ * D_ * 2);
    __bf16* hh    = (__bf16*)alloc((size_t)E_ * CAP_ * F_ * 2);
    float*  oexp  = (float*) alloc((size_t)E_ * CAP_ * D_ * 4);
    int*    eidx  = (int*)   alloc((size_t)N_ * 4);
    float*  gate  = (float*) alloc((size_t)N_ * 4);
    int*    slot  = (int*)   alloc((size_t)N_ * 4);
    float*  stats = (float*) alloc(32 * 4);

    // weight conversion to bf16
    k_cvt<<<2048, 256, 0, stream>>>(in_proj_w, wqkv, (long long)3 * D_ * D_);
    k_cvt<<<2048, 256, 0, stream>>>(out_proj_w, wout, (long long)D_ * D_);
    k_cvt<<<4096, 256, 0, stream>>>(w1, w1h, (long long)E_ * D_ * F_);
    k_cvt<<<4096, 256, 0, stream>>>(w2, w2h, (long long)E_ * F_ * D_);

    // LN1
    k_ln<<<N_ / 8, 256, 0, stream>>>(src, ln1_g, ln1_b, xh);

    // QKV projection: [N,3D] = xh[N,D] @ in_proj_w^T, bf16 out
    k_gemm<2><<<dim3(3 * D_ / 128, N_ / 64, 1), 256, 0, stream>>>(
        xh, wqkv, in_proj_b, nullptr, qkvh, N_, 3 * D_, D_, /*transB=*/1,
        0, 0, 0, 0);

    // attention
    k_attn<<<dim3(S_ / 128, H_, B_), 256, 0, stream>>>(qkvh, pad, attnh);

    // out-proj + residual: src1 = src + attnh @ out_proj_w^T + b
    k_gemm<1><<<dim3(D_ / 128, N_ / 64, 1), 256, 0, stream>>>(
        attnh, wout, out_proj_b, src, src1, N_, D_, D_, /*transB=*/1,
        0, 0, 0, 0);

    // LN2
    k_ln<<<N_ / 8, 256, 0, stream>>>(src1, ln2_g, ln2_b, x2h);

    // router + capacity scan
    k_zerof<<<1, 32, 0, stream>>>(stats, 32);
    k_router<<<N_ / 8, 256, 0, stream>>>(x2h, router_w, router_b, pad,
                                         eidx, gate, stats);
    k_scan<<<1, 32, 0, stream>>>(eidx, pad, slot);

    // dispatch (zero buffer first: dropped slots must be zero)
    k_zero16<<<2048, 256, 0, stream>>>((uint4*)disp,
                                       (long long)E_ * CAP_ * D_ / 8);
    k_dispatch<<<N_ / 8, 256, 0, stream>>>(x2h, slot, disp);

    // expert GEMM1: h = relu(disp @ w1[e] + b1[e]), bf16 out, batched over E
    k_gemm<3><<<dim3(F_ / 128, CAP_ / 64, E_), 256, 0, stream>>>(
        disp, w1h, b1, nullptr, hh, CAP_, F_, D_, /*transB=*/0,
        (long long)CAP_ * D_, (long long)D_ * F_, F_, (long long)CAP_ * F_);

    // expert GEMM2: o = h @ w2[e] + b2[e], fp32 out
    k_gemm<0><<<dim3(D_ / 128, CAP_ / 64, E_), 256, 0, stream>>>(
        hh, w2h, b2, nullptr, oexp, CAP_, D_, F_, /*transB=*/0,
        (long long)CAP_ * F_, (long long)F_ * D_, D_, (long long)CAP_ * D_);

    // gather + residual -> d_out, then losses
    k_final<<<N_ / 8, 256, 0, stream>>>(src1, oexp, slot, gate, out);
    k_loss<<<1, 32, 0, stream>>>(stats, out + (long long)N_ * D_);
}
